// LSTMGCN_58506044506787
// MI455X (gfx1250) — matrix operand order repaired and checked by reading, not compiled, not run on hardware
//
#include <hip/hip_runtime.h>
#include <hip/hip_bf16.h>

// ---------------------------------------------------------------------------
// GCN-LSTM for MI455X (gfx1250).  Dense GEMMs use v_wmma_f32_16x16x32_bf16
// with NO LDS: B (weights) is pre-packed in global memory in exact WMMA
// fragment order (2 x global_load_b128 per fragment), and A fragments are two
// contiguous 16B chunks of a row-major bf16 activation row (2 x b128).
// The edge scatter (dominant phase, ~16 GB of traffic) uses coalesced float4
// gathers + global float atomics.
// ---------------------------------------------------------------------------

typedef __attribute__((ext_vector_type(16))) __bf16 v16bf;
typedef __attribute__((ext_vector_type(8)))  float  v8f;

#define N_NODES  20000
#define N_EDGES  640000
#define HDIM     128
#define T_STEPS  12
#define OUT_DIM  64
#define NF       (N_NODES * HDIM)        // 2,560,000

#define BM 128                            // rows per block (8 waves x 16)
#define BN 64                             // cols per block (4 WMMA col tiles)

__device__ __forceinline__ unsigned short f32_to_bf16_rte(float f) {
  unsigned int u = __float_as_uint(f);
  u += 0x7FFFu + ((u >> 16) & 1u);       // round-to-nearest-even
  return (unsigned short)(u >> 16);
}
__device__ __forceinline__ float sigmoidf_(float x) {
  return 1.0f / (1.0f + __expf(-x));
}

// ---------------------------------------------------------------------------
// LDS-free WMMA GEMM:  C(M x Ncols, f32) = A(M x K, bf16 row-major)
//                                        @ Bp(K x Ncols, bf16, fragment-packed)
// Fragment packing of B (built once by pack kernels):
//   dword index = (((ct*Ktiles + kt)*32 + lane)*8 + v)
//   holds {B[k], B[k+1]} for col = ct*16 + lane%16, k = kt*32 + v*2 + (lane/16)*16
// A fragment per lane = A[row][kt*32 + hlf*16 .. +15] as two
// 16B chunks (kd = kt*16 + hlf*8 and kt*16 + hlf*8 + 4), same contiguous k order as Bp.
// Optional epilogue: aux[i] = gb2[col] + nself[row]*C[i]  (fused gcn_init).
// ---------------------------------------------------------------------------
template<int KK>
__global__ __launch_bounds__(256)
void gemm_bf16_wmma(const unsigned short* __restrict__ A,
                    const uint4* __restrict__ Bp,
                    const float* __restrict__ bias,   // nullable
                    float* __restrict__ C,
                    float* __restrict__ aux,          // nullable (gcn_init out)
                    const float* __restrict__ nself,
                    const float* __restrict__ gb2,
                    int M, int Ncols) {
  constexpr int KT = KK / 32;            // K tiles
  const int tid  = threadIdx.x;
  const int wave = tid >> 5;
  const int lane = tid & 31;
  const int hlf  = lane >> 4;
  const int ln16 = lane & 15;

  const int rowBase = blockIdx.x * BM + wave * 16;
  const int ctBase  = blockIdx.y * (BN / 16);

  // clamp A row for the ragged last block (stores are guarded)
  int arow = rowBase + ln16;
  if (arow >= M) arow = M - 1;
  const uint4* __restrict__ Aq = (const uint4*)(A + (size_t)arow * KK);

  v8f acc[4] = {};

#pragma unroll
  for (int kt = 0; kt < KT; ++kt) {
    union { v16bf v; uint4 q[2]; } afr;
    afr.q[0] = Aq[kt * 4 + hlf * 2];     // j = 0..3
    afr.q[1] = Aq[kt * 4 + hlf * 2 + 1]; // j = 4..7
#pragma unroll
    for (int j = 0; j < 4; ++j) {
      union { v16bf v; uint4 q[2]; } bfr;
      const uint4* bq = Bp + (((size_t)(ctBase + j) * KT + kt) * 32 + lane) * 2;
      bfr.q[0] = bq[0];
      bfr.q[1] = bq[1];
      acc[j] = __builtin_amdgcn_wmma_f32_16x16x32_bf16(
          false, afr.v, false, bfr.v, (short)0, acc[j], false, false);
    }
  }

  // C/D layout: VGPR p -> M = hlf*8 + p, N = lane%16
#pragma unroll
  for (int j = 0; j < 4; ++j) {
#pragma unroll
    for (int p = 0; p < 8; ++p) {
      int grow = rowBase + hlf * 8 + p;
      int gcol = (ctBase + j) * 16 + ln16;
      if (grow < M) {
        float v = acc[j][p];
        if (bias) v += bias[gcol];
        size_t idx = (size_t)grow * Ncols + gcol;
        C[idx] = v;
        if (aux) aux[idx] = gb2[gcol] + nself[grow] * v;
      }
    }
  }
}

// ---------------------------------------------------------------------------
// Weight packing into WMMA-fragment order (bf16 pairs per dword)
// ---------------------------------------------------------------------------
__global__ void pack_b_single_k(const float* __restrict__ W,   // K x Ncols f32
                                unsigned int* __restrict__ dst,
                                int K, int Ncols) {
  int idx = blockIdx.x * blockDim.x + threadIdx.x;
  if (idx >= (K * Ncols) / 2) return;
  int v    = idx & 7;
  int lane = (idx >> 3) & 31;
  int rest = idx >> 8;
  int kt   = rest % (K >> 5);
  int ct   = rest / (K >> 5);
  int col  = ct * 16 + (lane & 15);
  int k    = kt * 32 + v * 2 + (lane >> 4) * 16;
  unsigned int lo = f32_to_bf16_rte(W[(size_t)k * Ncols + col]);
  unsigned int hi = f32_to_bf16_rte(W[(size_t)(k + 1) * Ncols + col]);
  dst[idx] = lo | (hi << 16);
}
// Pack [wf|wi|wo|wc] (each K x 128 f32) as one K x 512 fragment-packed matrix.
__global__ void pack_b_gates_k(const float* __restrict__ wf, const float* __restrict__ wi,
                               const float* __restrict__ wo, const float* __restrict__ wc,
                               unsigned int* __restrict__ dst, int K) {
  int idx = blockIdx.x * blockDim.x + threadIdx.x;
  if (idx >= (K * 512) / 2) return;
  int v    = idx & 7;
  int lane = (idx >> 3) & 31;
  int rest = idx >> 8;
  int kt   = rest % (K >> 5);
  int ct   = rest / (K >> 5);
  int col  = ct * 16 + (lane & 15);
  int k    = kt * 32 + v * 2 + (lane >> 4) * 16;
  int sel  = col >> 7, f = col & 127;
  const float* src = (sel == 0) ? wf : (sel == 1) ? wi : (sel == 2) ? wo : wc;
  unsigned int lo = f32_to_bf16_rte(src[(size_t)k * 128 + f]);
  unsigned int hi = f32_to_bf16_rte(src[(size_t)(k + 1) * 128 + f]);
  dst[idx] = lo | (hi << 16);
}

// ---------------------------------------------------------------------------
// Graph-norm prep and pointwise kernels
// ---------------------------------------------------------------------------
__global__ void zero_f32_k(float* p, int n) {
  int i = blockIdx.x * blockDim.x + threadIdx.x;
  if (i < n) p[i] = 0.0f;
}
__global__ void zero_bf16_k(unsigned short* p, int n) {
  int i = blockIdx.x * blockDim.x + threadIdx.x;
  if (i < n) p[i] = 0;
}
__global__ void deg_k(const float* __restrict__ ea, const int* __restrict__ ei,
                      float* deg, int E) {
  int e = blockIdx.x * blockDim.x + threadIdx.x;
  if (e < E) atomicAdd(&deg[ei[E + e]], ea[e * 4 + 3]);
}
__global__ void dis_k(const float* __restrict__ deg, float* dis, float* norm_self, int n) {
  int i = blockIdx.x * blockDim.x + threadIdx.x;
  if (i < n) {
    float s = rsqrtf(deg[i] + 1.0f);     // +1 = self-loop weight; always > 0
    dis[i] = s;
    norm_self[i] = s * s;
  }
}
__global__ void norm_k(const int* __restrict__ ei, const float* __restrict__ ea,
                       const float* __restrict__ dis, float* norm, int E) {
  int e = blockIdx.x * blockDim.x + threadIdx.x;
  if (e < E) norm[e] = dis[ei[e]] * ea[e * 4 + 3] * dis[ei[E + e]];
}
// x is (N, 128, 12) row-major; extract timestep t as bf16 N x 128.
__global__ void extract_xt_k(const float* __restrict__ x, unsigned short* __restrict__ xt, int t) {
  int i = blockIdx.x * blockDim.x + threadIdx.x;
  if (i < NF) {
    int node = i >> 7, f = i & 127;
    xt[i] = f32_to_bf16_rte(x[(size_t)node * (128 * 12) + f * 12 + t]);
  }
}
// 32 lanes per edge, 4 feats/lane: out[col] += norm[e] * xw[row]
__global__ void gcn_scatter_k(const float* __restrict__ xw, const float* __restrict__ norm,
                              const int* __restrict__ ei, float* __restrict__ outp, int E) {
  int g = blockIdx.x * blockDim.x + threadIdx.x;
  int e = g >> 5, l = g & 31;
  if (e >= E) return;
  int   src = ei[e], dst = ei[E + e];
  float nm  = norm[e];
  const float4 xv = *(const float4*)(xw + (size_t)src * 128 + l * 4);
  float* op = outp + (size_t)dst * 128 + l * 4;
  atomicAdd(op + 0, nm * xv.x);
  atomicAdd(op + 1, nm * xv.y);
  atomicAdd(op + 2, nm * xv.z);
  atomicAdd(op + 3, nm * xv.w);
}
// zu = [xt | sigmoid(gcn) | h]  as bf16 (A operand of the fused gate GEMM)
__global__ void build_zu_k(const float* __restrict__ gcn, const unsigned short* __restrict__ xt,
                           const unsigned short* __restrict__ h, unsigned short* __restrict__ zu) {
  int i = blockIdx.x * blockDim.x + threadIdx.x;
  if (i < NF) {
    int node = i >> 7, f = i & 127;
    size_t zb = (size_t)node * 384;
    zu[zb + f]       = xt[i];
    zu[zb + 128 + f] = f32_to_bf16_rte(sigmoidf_(gcn[i]));
    zu[zb + 256 + f] = h[i];
  }
}
// gates: N x 512 f32 (f|i|o|c pre-activation, no bias yet)
__global__ void lstm_pointwise_k(const float* __restrict__ gates,
                                 const float* __restrict__ bf_, const float* __restrict__ bi_,
                                 const float* __restrict__ bo_, const float* __restrict__ bc_,
                                 float* __restrict__ c_st, unsigned short* __restrict__ h_bf) {
  int i = blockIdx.x * blockDim.x + threadIdx.x;
  if (i < NF) {
    int node = i >> 7, f = i & 127;
    const float* g = gates + (size_t)node * 512;
    float ft = sigmoidf_(g[f]       + bf_[f]);
    float it = sigmoidf_(g[128 + f] + bi_[f]);
    float ot = sigmoidf_(g[256 + f] + bo_[f]);
    float ct = tanhf    (g[384 + f] + bc_[f]);
    float cn = ft * c_st[i] + it * ct;
    c_st[i] = cn;
    h_bf[i] = f32_to_bf16_rte(ot * tanhf(cn));
  }
}

// ---------------------------------------------------------------------------
// Host orchestration
// ---------------------------------------------------------------------------
static inline int cdiv(int a, int b) { return (a + b - 1) / b; }

extern "C" void kernel_launch(void* const* d_in, const int* in_sizes, int n_in,
                              void* d_out, int out_size, void* d_ws, size_t ws_size,
                              hipStream_t stream) {
  const float* x  = (const float*)d_in[0];
  const float* ea = (const float*)d_in[1];
  const int*   ei = (const int*)d_in[2];
  const float* gw[2] = {(const float*)d_in[3],  (const float*)d_in[13]};
  const float* gb[2] = {(const float*)d_in[4],  (const float*)d_in[14]};
  const float* wg[2][4] = {
    {(const float*)d_in[5],  (const float*)d_in[7],  (const float*)d_in[9],  (const float*)d_in[11]},
    {(const float*)d_in[15], (const float*)d_in[17], (const float*)d_in[19], (const float*)d_in[21]}};
  const float* bg[2][4] = {
    {(const float*)d_in[6],  (const float*)d_in[8],  (const float*)d_in[10], (const float*)d_in[12]},
    {(const float*)d_in[16], (const float*)d_in[18], (const float*)d_in[20], (const float*)d_in[22]}};
  const float* ow = (const float*)d_in[23];
  const float* ob = (const float*)d_in[24];

  // ---- workspace bump allocator ----
  char* base = (char*)d_ws;
  size_t off = 0;
  auto bump = [&](size_t bytes) -> void* {
    void* p = base + off;
    off = (off + bytes + 255) & ~(size_t)255;
    return p;
  };
  unsigned int* gwp[2]   = {(unsigned int*)bump(128 * 128 * 2),
                            (unsigned int*)bump(128 * 128 * 2)};
  unsigned int* wcatp[2] = {(unsigned int*)bump(384 * 512 * 2),
                            (unsigned int*)bump(384 * 512 * 2)};
  unsigned int* owp      = (unsigned int*)bump(128 * 64 * 2);
  float* deg       = (float*)bump((size_t)N_NODES * 4);
  float* dis       = (float*)bump((size_t)N_NODES * 4);
  float* norm_self = (float*)bump((size_t)N_NODES * 4);
  float* norm      = (float*)bump((size_t)N_EDGES * 4);
  unsigned short* xt_bf   = (unsigned short*)bump((size_t)NF * 2);
  unsigned short* h_bf[2] = {(unsigned short*)bump((size_t)NF * 2),
                             (unsigned short*)bump((size_t)NF * 2)};
  float* c_st[2]  = {(float*)bump((size_t)NF * 4), (float*)bump((size_t)NF * 4)};
  float* xw       = (float*)bump((size_t)NF * 4);
  float* gcn_out  = (float*)bump((size_t)NF * 4);
  unsigned short* zu = (unsigned short*)bump((size_t)N_NODES * 384 * 2);
  float* gates    = (float*)bump((size_t)N_NODES * 512 * 4);

  const int TPB = 256;
  const int nf_blocks = cdiv(NF, TPB);

  // ---- weight packing (fragment order) ----
  pack_b_single_k<<<cdiv(128 * 128 / 2, TPB), TPB, 0, stream>>>(gw[0], gwp[0], 128, 128);
  pack_b_single_k<<<cdiv(128 * 128 / 2, TPB), TPB, 0, stream>>>(gw[1], gwp[1], 128, 128);
  pack_b_single_k<<<cdiv(128 * 64  / 2, TPB), TPB, 0, stream>>>(ow, owp, 128, 64);
  for (int l = 0; l < 2; ++l)
    pack_b_gates_k<<<cdiv(384 * 512 / 2, TPB), TPB, 0, stream>>>(
        wg[l][0], wg[l][1], wg[l][2], wg[l][3], wcatp[l], 384);

  // ---- graph normalization ----
  zero_f32_k<<<cdiv(N_NODES, TPB), TPB, 0, stream>>>(deg, N_NODES);
  deg_k<<<cdiv(N_EDGES, TPB), TPB, 0, stream>>>(ea, ei, deg, N_EDGES);
  dis_k<<<cdiv(N_NODES, TPB), TPB, 0, stream>>>(deg, dis, norm_self, N_NODES);
  norm_k<<<cdiv(N_EDGES, TPB), TPB, 0, stream>>>(ei, ea, dis, norm, N_EDGES);

  // ---- zero LSTM state ----
  for (int l = 0; l < 2; ++l) {
    zero_bf16_k<<<nf_blocks, TPB, 0, stream>>>(h_bf[l], NF);
    zero_f32_k<<<nf_blocks, TPB, 0, stream>>>(c_st[l], NF);
  }

  dim3 blk(TPB);
  dim3 g_gcn (cdiv(N_NODES, BM), HDIM / BN);      // 157 x 2
  dim3 g_gate(cdiv(N_NODES, BM), 512 / BN);       // 157 x 8
  dim3 g_outm(cdiv(N_NODES, BM), OUT_DIM / BN);   // 157 x 1
  const int scat_blocks = cdiv(N_EDGES * 32, TPB);

  // ---- time loop ----
  for (int t = 0; t < T_STEPS; ++t) {
    extract_xt_k<<<nf_blocks, TPB, 0, stream>>>(x, xt_bf, t);
    for (int l = 0; l < 2; ++l) {
      const unsigned short* xin = (l == 0) ? xt_bf : h_bf[0];
      // GCN matmul + fused init: xw = xin @ gw_l ; gcn_out = gb + nself*xw
      gemm_bf16_wmma<128><<<g_gcn, blk, 0, stream>>>(
          xin, (const uint4*)gwp[l], nullptr, xw,
          gcn_out, norm_self, gb[l], N_NODES, HDIM);
      // edge propagation: gcn_out[col] += norm[e] * xw[row]
      gcn_scatter_k<<<scat_blocks, TPB, 0, stream>>>(xw, norm, ei, gcn_out, N_EDGES);
      // zu = [xin | sigmoid(gcn_out) | h_l]  (bf16)
      build_zu_k<<<nf_blocks, TPB, 0, stream>>>(gcn_out, xin, h_bf[l], zu);
      // fused 4-gate GEMM: gates = zu @ [wf|wi|wo|wc]
      gemm_bf16_wmma<384><<<g_gate, blk, 0, stream>>>(
          zu, (const uint4*)wcatp[l], nullptr, gates,
          nullptr, nullptr, nullptr, N_NODES, 512);
      // LSTM state update (+ per-gate biases), h emitted as bf16
      lstm_pointwise_k<<<nf_blocks, TPB, 0, stream>>>(
          gates, bg[l][0], bg[l][1], bg[l][2], bg[l][3], c_st[l], h_bf[l]);
    }
  }

  // ---- output: d_out = h1 @ ow + ob ----
  gemm_bf16_wmma<128><<<g_outm, blk, 0, stream>>>(
      h_bf[1], (const uint4*)owp, ob, (float*)d_out,
      nullptr, nullptr, nullptr, N_NODES, OUT_DIM);
}
